// NeuralQueryTransformer_53584011985577
// MI455X (gfx1250) — compile-verified, matrix operand order checked by reading, not run on hardware
//
#include <hip/hip_runtime.h>
#include <hip/hip_bf16.h>

// ---- model dims ----
#define V_  32000
#define E_  256
#define H_  512
#define DH_ 1024
#define NH_ 4
#define HD_ 256
#define B_  64
#define S_  128
#define T_  50

typedef __bf16 bf16_t;
typedef __attribute__((ext_vector_type(16))) __bf16 v16bf;
typedef __attribute__((ext_vector_type(8)))  float  v8f;

__device__ __forceinline__ float sigmoidf_(float x) { return 1.0f / (1.0f + __expf(-x)); }

// ---------------- conversion / init kernels ----------------
__global__ void k_cvt_bf16(const float* __restrict__ s, bf16_t* __restrict__ d, int n) {
    int i = blockIdx.x * blockDim.x + threadIdx.x;
    if (i < n) d[i] = (bf16_t)s[i];
}

__global__ void k_zero_state(bf16_t* __restrict__ hA, bf16_t* __restrict__ hB,
                             float* __restrict__ c, int n) {
    int i = blockIdx.x * blockDim.x + threadIdx.x;
    if (i < n) { hA[i] = (bf16_t)0.0f; hB[i] = (bf16_t)0.0f; c[i] = 0.0f; }
}

// copy encoder final (h,c) [64,512] into decoder init buffers [64,1024] at coloff
__global__ void k_copy_state(const bf16_t* __restrict__ hs, const float* __restrict__ cs,
                             bf16_t* __restrict__ hd, float* __restrict__ cd, int coloff) {
    int i = blockIdx.x * blockDim.x + threadIdx.x;
    if (i < B_ * H_) {
        int b = i >> 9, j = i & 511;
        hd[b * DH_ + coloff + j] = hs[i];
        cd[b * DH_ + coloff + j] = cs[i];
    }
}

__global__ void k_init_tok(int* __restrict__ tok) {
    if (threadIdx.x < B_) tok[threadIdx.x] = 0;
}

// X0[s*B+b][e] = embT[input_ids[b][s]][e]   (bf16 gather for the whole sequence)
__global__ void k_embed_enc(const int* __restrict__ ids, const bf16_t* __restrict__ embT,
                            bf16_t* __restrict__ X) {
    int i = blockIdx.x * blockDim.x + threadIdx.x;
    if (i < S_ * B_ * E_) {
        int e = i & (E_ - 1);
        int row = i >> 8;            // s*B + b
        int s = row >> 6, b = row & 63;
        int tok = ids[b * S_ + s];
        X[i] = embT[(long)tok * E_ + e];
    }
}

__global__ void k_embed_tok(const int* __restrict__ tok, const bf16_t* __restrict__ embT,
                            bf16_t* __restrict__ x) {
    int i = blockIdx.x * blockDim.x + threadIdx.x;
    if (i < B_ * E_) {
        int e = i & (E_ - 1), b = i >> 8;
        x[i] = embT[(long)tok[b] * E_ + e];
    }
}

// ---------------- generic WMMA GEMM: C[M,N] = A[M,K] @ W[N,K]^T + bias ----------------
// One wave computes a 16x64 strip (4 N-tiles): 1 A-fragment load feeds 4 independent
// WMMAs (4 B-fragments), quartering A traffic and keeping the XDL pipe busy.
// A,W bf16 row-major (K contiguous), C fp32 (ldc row stride), optional bf16 copy.
__global__ __launch_bounds__(32) void k_gemm_bf16(
    const bf16_t* __restrict__ A, int lda,
    const bf16_t* __restrict__ W, int ldw,
    const float* __restrict__ bias, int K,
    float* __restrict__ C, long ldc,
    bf16_t* __restrict__ Cbf, int ldcbf)
{
    const int lane = threadIdx.x;
    const int lm = lane & 15, lh = lane >> 4;
    const int n0 = blockIdx.x * 64, m0 = blockIdx.y * 16;
    v8f acc[4];
#pragma unroll
    for (int nb = 0; nb < 4; ++nb) {
        float bv = bias ? bias[n0 + nb * 16 + lm] : 0.0f;
#pragma unroll
        for (int r = 0; r < 8; ++r) acc[nb][r] = bv;
    }
    const bf16_t* ap = A + (long)(m0 + lm) * lda + lh * 16;
    const bf16_t* wp = W + (long)(n0 + lm) * ldw + lh * 16;
    const long wstep = (long)16 * ldw;           // N-tile stride in W
    for (int kb = 0; kb < K; kb += 32) {
        __builtin_prefetch((const void*)(wp + kb + 32), 0, 1);  // next k-block of weight stream
        v16bf a = *(const v16bf*)(ap + kb);
#pragma unroll
        for (int nb = 0; nb < 4; ++nb) {
            v16bf b = *(const v16bf*)(wp + (long)nb * wstep + kb);
            acc[nb] = __builtin_amdgcn_wmma_f32_16x16x32_bf16(false, a, false, b, (short)0, acc[nb], false, false);
        }
    }
    const int rb = m0 + lh * 8;
#pragma unroll
    for (int nb = 0; nb < 4; ++nb) {
        const int col = n0 + nb * 16 + lm;
#pragma unroll
        for (int r = 0; r < 8; ++r) {
            C[(long)(rb + r) * ldc + col] = acc[nb][r];
            if (Cbf) Cbf[(long)(rb + r) * ldcbf + col] = (bf16_t)acc[nb][r];
        }
    }
}

// ---------------- fused LSTM step: gates GEMM (WMMA) + cell pointwise in registers ----
// gates = X@Wih^T + Hprev@Whh^T + bih + bhh; wave owns the 4 gate tiles (i|f|g|o) for
// one (batch-tile, hidden-tile), so c is wave-exclusive and updated in place.
__global__ __launch_bounds__(32) void k_lstm_step(
    const bf16_t* __restrict__ X, int ldx, int K1,
    const bf16_t* __restrict__ Wih,
    const bf16_t* __restrict__ Hprev,
    const bf16_t* __restrict__ Whh,
    const float* __restrict__ bih, const float* __restrict__ bhh,
    float* __restrict__ Cst,
    bf16_t* __restrict__ Hout,
    bf16_t* __restrict__ Xdst, long ldxd,
    int Hn)
{
    const int lane = threadIdx.x;
    const int lm = lane & 15, lh = lane >> 4;
    const int j0 = blockIdx.x * 16;
    const int m0 = blockIdx.y * 16;
    v8f acc[4];
#pragma unroll
    for (int g = 0; g < 4; ++g) {
        float bv = bih[g * Hn + j0 + lm] + bhh[g * Hn + j0 + lm];
#pragma unroll
        for (int r = 0; r < 8; ++r) acc[g][r] = bv;
    }
    {   // input-projection contribution
        const bf16_t* ap = X + (long)(m0 + lm) * ldx + lh * 16;
        for (int kb = 0; kb < K1; kb += 32) {
            v16bf a = *(const v16bf*)(ap + kb);
#pragma unroll
            for (int g = 0; g < 4; ++g) {
                v16bf b = *(const v16bf*)(Wih + (long)(g * Hn + j0 + lm) * K1 + kb + lh * 16);
                acc[g] = __builtin_amdgcn_wmma_f32_16x16x32_bf16(false, a, false, b, (short)0, acc[g], false, false);
            }
        }
    }
    {   // recurrent contribution (K2 == Hn)
        const bf16_t* ap = Hprev + (long)(m0 + lm) * Hn + lh * 16;
        for (int kb = 0; kb < Hn; kb += 32) {
            v16bf a = *(const v16bf*)(ap + kb);
#pragma unroll
            for (int g = 0; g < 4; ++g) {
                v16bf b = *(const v16bf*)(Whh + (long)(g * Hn + j0 + lm) * Hn + kb + lh * 16);
                acc[g] = __builtin_amdgcn_wmma_f32_16x16x32_bf16(false, a, false, b, (short)0, acc[g], false, false);
            }
        }
    }
    // pointwise LSTM cell, fragment layout: row = m0 + r + 8*lh, col = j0 + lm
    const int col = j0 + lm;
    const int rb = m0 + lh * 8;
#pragma unroll
    for (int r = 0; r < 8; ++r) {
        int bi = rb + r;
        float iv = sigmoidf_(acc[0][r]);
        float fv = sigmoidf_(acc[1][r]);
        float gv = tanhf(acc[2][r]);
        float ov = sigmoidf_(acc[3][r]);
        long ci = (long)bi * Hn + col;
        float c2 = fv * Cst[ci] + iv * gv;
        Cst[ci] = c2;
        float h2 = ov * tanhf(c2);
        Hout[ci] = (bf16_t)h2;
        if (Xdst) Xdst[(long)bi * ldxd + col] = (bf16_t)h2;
    }
}

// ---------------- multi-head attention (fp32 VALU; tiny FLOP count) -------------------
// one block per (head, batch); 128 threads = one per source position
__global__ void k_attn(const float* __restrict__ Q, const float* __restrict__ Km,
                       const float* __restrict__ Vm, const int* __restrict__ mask,
                       bf16_t* __restrict__ ctx)
{
    __shared__ float sc[S_];
    __shared__ float red[S_];
    int h = blockIdx.x, b = blockIdx.y, s = threadIdx.x;
    const float* q  = Q + b * DH_ + h * HD_;
    const float* kr = Km + (long)(s * B_ + b) * DH_ + h * HD_;
    float d = 0.0f;
    for (int i = 0; i < HD_; i += 4)
        d += q[i] * kr[i] + q[i + 1] * kr[i + 1] + q[i + 2] * kr[i + 2] + q[i + 3] * kr[i + 3];
    d *= 0.0625f;                       // 1/sqrt(HD)
    if (mask[b * S_ + s] == 0) d = -1e9f;
    sc[s] = d; red[s] = d; __syncthreads();
    for (int off = 64; off > 0; off >>= 1) { if (s < off) red[s] = fmaxf(red[s], red[s + off]); __syncthreads(); }
    float mx = red[0]; __syncthreads();
    float e = __expf(sc[s] - mx);
    sc[s] = e; red[s] = e; __syncthreads();
    for (int off = 64; off > 0; off >>= 1) { if (s < off) red[s] += red[s + off]; __syncthreads(); }
    float inv = 1.0f / red[0]; __syncthreads();
    sc[s] *= inv; __syncthreads();
    // ctx: each thread accumulates dims s and s+128 of this head
    float a0 = 0.0f, a1 = 0.0f;
    for (int ss = 0; ss < S_; ++ss) {
        float w = sc[ss];
        const float* vr = Vm + (long)(ss * B_ + b) * DH_ + h * HD_;
        a0 += w * vr[s];
        a1 += w * vr[s + 128];
    }
    ctx[b * DH_ + h * HD_ + s]       = (bf16_t)a0;
    ctx[b * DH_ + h * HD_ + s + 128] = (bf16_t)a1;
}

// ---------------- argmax over vocab (greedy feedback), lowest-index tie-break ---------
__global__ void k_argmax(const float* __restrict__ logits, long ldl, int* __restrict__ tok) {
    __shared__ float sv[256];
    __shared__ int   si[256];
    int b = blockIdx.x, t = threadIdx.x;
    const float* row = logits + (long)b * ldl;
    float best = -3.4e38f; int bi = 0;
    for (int v = t; v < V_; v += 256) {
        float x = row[v];
        if (x > best) { best = x; bi = v; }
    }
    sv[t] = best; si[t] = bi; __syncthreads();
    for (int off = 128; off > 0; off >>= 1) {
        if (t < off) {
            float xo = sv[t + off]; int io = si[t + off];
            if (xo > sv[t] || (xo == sv[t] && io < si[t])) { sv[t] = xo; si[t] = io; }
        }
        __syncthreads();
    }
    if (t == 0) tok[b] = si[0];
}

// =====================================================================================
extern "C" void kernel_launch(void* const* d_in, const int* in_sizes, int n_in,
                              void* d_out, int out_size, void* d_ws, size_t ws_size,
                              hipStream_t stream) {
    (void)in_sizes; (void)n_in; (void)out_size; (void)ws_size;

    const int*   input_ids  = (const int*)d_in[0];
    const int*   input_mask = (const int*)d_in[1];
    const float* emb        = (const float*)d_in[2];
    // encoder params: scans ordered (l0,f),(l0,b),(l1,f),(l1,b), each 4 arrays
    const float *e_w_ih[4], *e_w_hh[4], *e_b_ih[4], *e_b_hh[4];
    for (int k = 0; k < 4; ++k) {
        e_w_ih[k] = (const float*)d_in[3 + 4 * k + 0];
        e_w_hh[k] = (const float*)d_in[3 + 4 * k + 1];
        e_b_ih[k] = (const float*)d_in[3 + 4 * k + 2];
        e_b_hh[k] = (const float*)d_in[3 + 4 * k + 3];
    }
    const float* d_w_ih[2] = { (const float*)d_in[19], (const float*)d_in[23] };
    const float* d_w_hh[2] = { (const float*)d_in[20], (const float*)d_in[24] };
    const float* d_b_ih[2] = { (const float*)d_in[21], (const float*)d_in[25] };
    const float* d_b_hh[2] = { (const float*)d_in[22], (const float*)d_in[26] };
    const float* attn_in_w  = (const float*)d_in[27];
    const float* attn_in_b  = (const float*)d_in[28];
    const float* attn_out_w = (const float*)d_in[29];
    const float* attn_out_b = (const float*)d_in[30];
    const float* proj_w     = (const float*)d_in[31];
    const float* proj_b     = (const float*)d_in[32];
    float* out = (float*)d_out;

    // ---- workspace allocator (256B aligned) ----
    char*  base = (char*)d_ws;
    size_t off  = 0;
    auto alloc = [&](size_t bytes) -> char* {
        size_t a = (off + 255) & ~(size_t)255;
        off = a + bytes;
        return base + a;
    };

    // bf16 weights
    bf16_t* embT   = (bf16_t*)alloc((size_t)V_ * E_ * 2);
    bf16_t* ewih[4]; bf16_t* ewhh[4];
    const size_t ewih_n[4] = { 2048u*256, 2048u*256, 2048u*1024, 2048u*1024 };
    for (int k = 0; k < 4; ++k) {
        ewih[k] = (bf16_t*)alloc(ewih_n[k] * 2);
        ewhh[k] = (bf16_t*)alloc((size_t)2048 * 512 * 2);
    }
    bf16_t* dwih[2]; bf16_t* dwhh[2];
    const size_t dwih_n[2] = { 4096u*256, 4096u*1024 };
    for (int l = 0; l < 2; ++l) {
        dwih[l] = (bf16_t*)alloc(dwih_n[l] * 2);
        dwhh[l] = (bf16_t*)alloc((size_t)4096 * 1024 * 2);
    }
    bf16_t* awin  = (bf16_t*)alloc((size_t)3 * DH_ * DH_ * 2);
    bf16_t* awout = (bf16_t*)alloc((size_t)DH_ * DH_ * 2);
    bf16_t* pw    = (bf16_t*)alloc((size_t)V_ * DH_ * 2);

    // activations
    bf16_t* X0 = (bf16_t*)alloc((size_t)S_ * B_ * E_ * 2);
    bf16_t* X1 = (bf16_t*)alloc((size_t)S_ * B_ * DH_ * 2);
    bf16_t* X2 = (bf16_t*)alloc((size_t)S_ * B_ * DH_ * 2);
    bf16_t* hA = (bf16_t*)alloc((size_t)B_ * H_ * 2);
    bf16_t* hB = (bf16_t*)alloc((size_t)B_ * H_ * 2);
    float*  cscan = (float*)alloc((size_t)B_ * H_ * 4);
    bf16_t* dh[2][2]; float* dc[2];
    for (int l = 0; l < 2; ++l) {
        dh[l][0] = (bf16_t*)alloc((size_t)B_ * DH_ * 2);
        dh[l][1] = (bf16_t*)alloc((size_t)B_ * DH_ * 2);
        dc[l]    = (float*) alloc((size_t)B_ * DH_ * 4);
    }
    float*  Kf    = (float*) alloc((size_t)S_ * B_ * DH_ * 4);
    float*  Vf    = (float*) alloc((size_t)S_ * B_ * DH_ * 4);
    float*  Qf    = (float*) alloc((size_t)B_ * DH_ * 4);
    bf16_t* ctxbf = (bf16_t*)alloc((size_t)B_ * DH_ * 2);
    float*  aof   = (float*) alloc((size_t)B_ * DH_ * 4);
    bf16_t* aobf  = (bf16_t*)alloc((size_t)B_ * DH_ * 2);
    bf16_t* xdec  = (bf16_t*)alloc((size_t)B_ * E_ * 2);
    int*    tok   = (int*)   alloc((size_t)B_ * 4);

    auto cvt = [&](const float* s, bf16_t* d, size_t n) {
        k_cvt_bf16<<<(unsigned)((n + 255) / 256), 256, 0, stream>>>(s, d, (int)n);
    };

    // ---- weight conversions (bf16, L2-resident working set) ----
    cvt(emb, embT, (size_t)V_ * E_);
    for (int k = 0; k < 4; ++k) { cvt(e_w_ih[k], ewih[k], ewih_n[k]); cvt(e_w_hh[k], ewhh[k], (size_t)2048 * 512); }
    for (int l = 0; l < 2; ++l) { cvt(d_w_ih[l], dwih[l], dwih_n[l]); cvt(d_w_hh[l], dwhh[l], (size_t)4096 * 1024); }
    cvt(attn_in_w, awin, (size_t)3 * DH_ * DH_);
    cvt(attn_out_w, awout, (size_t)DH_ * DH_);
    cvt(proj_w, pw, (size_t)V_ * DH_);

    // ---- encoder embedding gather ----
    k_embed_enc<<<(S_ * B_ * E_) / 256, 256, 0, stream>>>(input_ids, embT, X0);

    // ---- encoder: 4 sequential scans (l0 fwd, l0 bwd, l1 fwd, l1 bwd) ----
    for (int sc = 0; sc < 4; ++sc) {
        const bf16_t* Xin = (sc < 2) ? X0 : X1;
        int ldx = (sc < 2) ? E_ : DH_;
        bf16_t* Xout = (sc < 2) ? X1 : X2;
        int coloff = (sc & 1) ? H_ : 0;
        int l = sc >> 1;
        k_zero_state<<<(B_ * H_ + 255) / 256, 256, 0, stream>>>(hA, hB, cscan, B_ * H_);
        for (int stp = 0; stp < S_; ++stp) {
            int t = (sc & 1) ? (S_ - 1 - stp) : stp;
            bf16_t* hin  = (stp & 1) ? hB : hA;
            bf16_t* hout = (stp & 1) ? hA : hB;
            k_lstm_step<<<dim3(H_ / 16, B_ / 16), 32, 0, stream>>>(
                Xin + (long)t * B_ * ldx, ldx, ldx, ewih[sc],
                hin, ewhh[sc], e_b_ih[sc], e_b_hh[sc],
                cscan, hout,
                Xout + (long)t * B_ * DH_ + coloff, (long)DH_, H_);
        }
        // 128 steps -> final h in hA; final c in cscan. Becomes decoder init state.
        k_copy_state<<<(B_ * H_ + 255) / 256, 256, 0, stream>>>(hA, cscan, dh[l][0], dc[l], coloff);
    }

    // ---- loop-invariant K/V projections over encoder output (whole-chip WMMA GEMM) ----
    k_gemm_bf16<<<dim3(DH_ / 64, (S_ * B_) / 16), 32, 0, stream>>>(
        X2, DH_, awin + (size_t)DH_ * DH_, DH_, attn_in_b + DH_, DH_, Kf, DH_, nullptr, 0);
    k_gemm_bf16<<<dim3(DH_ / 64, (S_ * B_) / 16), 32, 0, stream>>>(
        X2, DH_, awin + (size_t)2 * DH_ * DH_, DH_, attn_in_b + 2 * DH_, DH_, Vf, DH_, nullptr, 0);

    // ---- greedy decode, T steps ----
    k_init_tok<<<1, 64, 0, stream>>>(tok);
    for (int t = 0; t < T_; ++t) {
        int pi = t & 1, po = pi ^ 1;
        k_embed_tok<<<(B_ * E_ + 255) / 256, 256, 0, stream>>>(tok, embT, xdec);
        k_lstm_step<<<dim3(DH_ / 16, B_ / 16), 32, 0, stream>>>(
            xdec, E_, E_, dwih[0], dh[0][pi], dwhh[0], d_b_ih[0], d_b_hh[0],
            dc[0], dh[0][po], nullptr, 0, DH_);
        k_lstm_step<<<dim3(DH_ / 16, B_ / 16), 32, 0, stream>>>(
            dh[0][po], DH_, DH_, dwih[1], dh[1][pi], dwhh[1], d_b_ih[1], d_b_hh[1],
            dc[1], dh[1][po], nullptr, 0, DH_);
        // Q projection
        k_gemm_bf16<<<dim3(DH_ / 64, B_ / 16), 32, 0, stream>>>(
            dh[1][po], DH_, awin, DH_, attn_in_b, DH_, Qf, DH_, nullptr, 0);
        // attention
        k_attn<<<dim3(NH_, B_), S_, 0, stream>>>(Qf, Kf, Vf, input_mask, ctxbf);
        // attention output projection (also emit bf16 for logits GEMM)
        k_gemm_bf16<<<dim3(DH_ / 64, B_ / 16), 32, 0, stream>>>(
            ctxbf, DH_, awout, DH_, attn_out_b, DH_, aof, DH_, aobf, DH_);
        // logits -> directly into d_out[b, t, :]; proj_w (bf16, 65MB) stays hot in L2
        k_gemm_bf16<<<dim3(V_ / 64, B_ / 16), 32, 0, stream>>>(
            aobf, DH_, pw, DH_, proj_b, DH_, out + (long)t * V_, (long)T_ * V_, nullptr, 0);
        // greedy feedback
        k_argmax<<<B_, 256, 0, stream>>>(out + (long)t * V_, (long)T_ * V_, tok);
    }
}
